// DualEdgeEGNN_67001489818047
// MI455X (gfx1250) — compile-verified, matrix operand order checked by reading, not run on hardware
//
#include <hip/hip_runtime.h>
#include <hip/hip_bf16.h>
#include <math.h>

// ---------------------------------------------------------------------------
// Problem constants (match reference)
// ---------------------------------------------------------------------------
#define NNODES 10000
#define NEDGES 160000
#define NGRAPH 128
#define HID    128
#define NUM_ATOM 16

typedef __attribute__((ext_vector_type(16))) _Float16 v16h;
typedef __attribute__((ext_vector_type(8)))  _Float16 v8h;
typedef __attribute__((ext_vector_type(8)))  float    v8f;

// ---------------------------------------------------------------------------
// WMMA fragment layout helpers (CDNA5 wave32, V_WMMA_F32_16X16X32_F16)
//   A 16x32 f16 : lane row = lane&15 ; k = h + 8*(h>=8) + 8*(lane>=16)
//   B 32x16 f16 : lane col = lane&15 ; same k mapping
//   C/D 16x16 f32: row = r + 8*(lane>=16) ; col = lane&15
// All activations in LDS and all weights in global are stored pre-swizzled in
// "fragment layout": [ktile][lane][h] so a lane reads its 16-half fragment as
// one contiguous 32-byte vector (2x b128 loads).
// ---------------------------------------------------------------------------
__device__ __forceinline__ int frag_m(int r, int lane) {
  return r + ((lane & 16) ? 8 : 0);
}
// position (in halves) of logical element (row r, k=f) in fragment-layout buffer
__device__ __forceinline__ int aidx(int r, int f) {
  const int kt = f >> 5, kk = f & 31;
  const int g = (kk >> 3) & 1;
  const int h = (kk & 7) + ((kk & 16) >> 1);
  return (((kt << 5) + (g << 4) + r) << 4) + h;
}
// store 8 consecutive logical k-values (f0 % 8 == 0) for row r — contiguous
__device__ __forceinline__ void store_a8(_Float16* __restrict__ buf, int r, int f0,
                                         const float* v) {
  v8h o;
#pragma unroll
  for (int i = 0; i < 8; ++i) o[i] = (_Float16)v[i];
  *(v8h*)(buf + aidx(r, f0)) = o;
}

// One 16x16 output tile: C = A(16 x 32*ktiles, frag LDS) * Wf(frag global) + bias
__device__ __forceinline__ v8f mma_tile(const _Float16* __restrict__ A,
                                        const _Float16* __restrict__ Wf, int NT,
                                        const float* __restrict__ bias, int nreal,
                                        int ktiles, int nt, int lane) {
  const int col = nt * 16 + (lane & 15);
  const float bv = (col < nreal) ? bias[col] : 0.0f;
  v8f acc;
#pragma unroll
  for (int r = 0; r < 8; ++r) acc[r] = bv;
  for (int kt = 0; kt < ktiles; ++kt) {
    v16h a = *(const v16h*)(A + (((kt << 5) + lane) << 4));
    v16h b = *(const v16h*)(Wf + ((((size_t)kt * NT + nt) << 5) + lane) * 16);
    acc = __builtin_amdgcn_wmma_f32_16x16x32_f16(false, a, false, b,
                                                 (short)0, acc, false, false);
  }
  return acc;
}

__device__ __forceinline__ float silu_f(float x) { return x / (1.0f + __expf(-x)); }
__device__ __forceinline__ float sigmoid_f(float x) { return 1.0f / (1.0f + __expf(-x)); }

// ---------------------------------------------------------------------------
// Weight conversion: f32 [K,N] row-major -> f16 fragment layout [KT][NT][lane][h]
// ---------------------------------------------------------------------------
__global__ void convert_frag_kernel(const float* __restrict__ src, _Float16* __restrict__ dst,
                                    int K, int Nn, int KT, int NT) {
  int i = blockIdx.x * 256 + threadIdx.x;
  if (i >= KT * NT * 512) return;
  const int h = i & 15;
  const int lane = (i >> 4) & 31;
  const int tile = i >> 9;
  const int nt = tile % NT, kt = tile / NT;
  const int k = (kt << 5) + (h & 7) + ((h & 8) << 1) + ((lane & 16) >> 1);
  const int n = (nt << 4) + (lane & 15);
  dst[i] = (k < K && n < Nn) ? (_Float16)src[(size_t)k * Nn + n] : (_Float16)0.0f;
}

__global__ void zero_f32_kernel(float* p, int n) {
  int i = blockIdx.x * 256 + threadIdx.x;
  if (i < n) p[i] = 0.0f;
}

__global__ void copy_pos_kernel(const float* __restrict__ src, float* __restrict__ dst, int n) {
  int i = blockIdx.x * 256 + threadIdx.x;
  if (i < n) dst[i] = src[i];
}

__global__ void aux_out_kernel(const int* __restrict__ eidx, const int* __restrict__ etype,
                               const float* __restrict__ elen, float* __restrict__ out) {
  int e = blockIdx.x * 256 + threadIdx.x;
  if (e >= NEDGES) return;
  out[(size_t)NEDGES * 1 + e] = (float)eidx[e];
  out[(size_t)NEDGES * 2 + e] = (float)eidx[NEDGES + e];
  out[(size_t)NEDGES * 3 + e] = (float)etype[e];
  out[(size_t)NEDGES * 4 + e] = elen[e];
  out[(size_t)NEDGES * 5 + e] = (etype[e] == 0) ? 1.0f : 0.0f;
}

// ---------------------------------------------------------------------------
// Timestep-embedding MLP: 16 graphs / block
// ---------------------------------------------------------------------------
__global__ void __launch_bounds__(128)
temb_kernel(const int* __restrict__ tstep,
            const _Float16* w_t0, const float* b_t0,
            const _Float16* w_t1, const float* b_t1,
            const _Float16* w_tp, const float* b_tp,
            float* __restrict__ temb) {
  __shared__ __align__(32) _Float16 x0[16 * 128];
  __shared__ __align__(32) _Float16 x1[16 * 512];
  __shared__ __align__(32) _Float16 x2[16 * 512];
  const int tid = threadIdx.x, wave = tid >> 5, lane = tid & 31;
  const int g0 = blockIdx.x * 16;

  for (int gi = tid; gi < 16 * 16; gi += 128) {       // 16 rows x 16 groups of 8
    int r = gi >> 4, f0 = (gi & 15) << 3;
    float t = (float)tstep[g0 + r];
    float v[8];
#pragma unroll
    for (int i = 0; i < 8; ++i) {
      int f = f0 + i;
      float freq = __expf(-9.210340372f * (float)(f & 63) / 63.0f); // ln(1e4)
      float a = t * freq;
      v[i] = (f < 64) ? __sinf(a) : __cosf(a);
    }
    store_a8(x0, r, f0, v);
  }
  __syncthreads();
  for (int nt = wave; nt < 32; nt += 4) {
    v8f acc = mma_tile(x0, w_t0, 32, b_t0, 512, 4, nt, lane);
    int col = nt * 16 + (lane & 15);
#pragma unroll
    for (int r = 0; r < 8; ++r)
      x1[aidx(frag_m(r, lane), col)] = (_Float16)fmaxf(acc[r], 0.0f);
  }
  __syncthreads();
  for (int nt = wave; nt < 32; nt += 4) {
    v8f acc = mma_tile(x1, w_t1, 32, b_t1, 512, 16, nt, lane);
    int col = nt * 16 + (lane & 15);
#pragma unroll
    for (int r = 0; r < 8; ++r)
      x2[aidx(frag_m(r, lane), col)] = (_Float16)fmaxf(acc[r], 0.0f); // relu before proj
  }
  __syncthreads();
  for (int nt = wave; nt < 8; nt += 4) {
    v8f acc = mma_tile(x2, w_tp, 8, b_tp, 128, 16, nt, lane);
    int col = nt * 16 + (lane & 15);
#pragma unroll
    for (int r = 0; r < 8; ++r)
      temb[(size_t)(g0 + frag_m(r, lane)) * 128 + col] = acc[r];
  }
}

// ---------------------------------------------------------------------------
// Edge encoder: eattr = (relu(len*w0+b0) @ W1 + b1)*bond_emb[et] + temb[batch[row]]
// ---------------------------------------------------------------------------
__global__ void __launch_bounds__(128)
edge_attr_kernel(const float* __restrict__ elen, const int* __restrict__ etype,
                 const int* __restrict__ eidx, const int* __restrict__ batch,
                 const float* __restrict__ w_m0, const float* __restrict__ b_m0,
                 const _Float16* w_m1, const float* b_m1,
                 const float* __restrict__ bond_emb, const float* __restrict__ temb,
                 _Float16* __restrict__ eattr) {
  __shared__ __align__(32) _Float16 x_s[16 * 128];
  const int tid = threadIdx.x, wave = tid >> 5, lane = tid & 31;
  const int e0 = blockIdx.x * 16;
  for (int gi = tid; gi < 16 * 16; gi += 128) {
    int r = gi >> 4, f0 = (gi & 15) << 3;
    float len = elen[e0 + r];
    float v[8];
#pragma unroll
    for (int i = 0; i < 8; ++i) v[i] = fmaxf(len * w_m0[f0 + i] + b_m0[f0 + i], 0.0f);
    store_a8(x_s, r, f0, v);
  }
  __syncthreads();
  for (int nt = wave; nt < 8; nt += 4) {
    v8f acc = mma_tile(x_s, w_m1, 8, b_m1, 128, 4, nt, lane);
    int col = nt * 16 + (lane & 15);
#pragma unroll
    for (int r = 0; r < 8; ++r) {
      int e = e0 + frag_m(r, lane);
      int et = etype[e];
      int bg = batch[eidx[e]];
      float v = acc[r] * bond_emb[(size_t)et * 128 + col] + temb[(size_t)bg * 128 + col];
      eattr[(size_t)e * 128 + col] = (_Float16)v;
    }
  }
}

// ---------------------------------------------------------------------------
// Node embedding: h = [atom_type, temb[batch]] @ W_embed + b
// ---------------------------------------------------------------------------
__global__ void __launch_bounds__(128)
embed_kernel(const float* __restrict__ atom_type, const int* __restrict__ batch,
             const float* __restrict__ temb,
             const _Float16* w_emb, const float* b_emb,
             float* __restrict__ h) {
  __shared__ __align__(32) _Float16 in_s[16 * 160];
  const int tid = threadIdx.x, wave = tid >> 5, lane = tid & 31;
  const int n0 = blockIdx.x * 16;
  for (int gi = tid; gi < 16 * 20; gi += 128) {      // 160 features = 20 groups
    int r = gi / 20, f0 = (gi % 20) << 3;
    float v[8];
#pragma unroll
    for (int i = 0; i < 8; ++i) v[i] = 0.0f;
    if (f0 < NUM_ATOM) {
#pragma unroll
      for (int i = 0; i < 8; ++i) v[i] = atom_type[(size_t)(n0 + r) * NUM_ATOM + f0 + i];
    } else if (f0 < NUM_ATOM + HID) {
      const float* tp = temb + (size_t)batch[n0 + r] * 128 + (f0 - NUM_ATOM);
#pragma unroll
      for (int i = 0; i < 8; ++i) v[i] = tp[i];
    }
    store_a8(in_s, r, f0, v);
  }
  __syncthreads();
  for (int nt = wave; nt < 8; nt += 4) {
    v8f acc = mma_tile(in_s, w_emb, 8, b_emb, 128, 5, nt, lane);
    int col = nt * 16 + (lane & 15);
#pragma unroll
    for (int r = 0; r < 8; ++r)
      h[(size_t)(n0 + frag_m(r, lane)) * 128 + col] = acc[r];
  }
}

// ---------------------------------------------------------------------------
// Fused EGNN edge-message kernel: 16 edges / block, 4 waves, all LDS-resident.
// ---------------------------------------------------------------------------
__global__ void __launch_bounds__(128)
edge_msg_kernel(const int* __restrict__ eidx,
                const float* __restrict__ posc, const float* __restrict__ h,
                const _Float16* __restrict__ eattr,
                const _Float16* w_e0, const float* b_e0,
                const _Float16* w_e1, const float* b_e1,
                const float* __restrict__ w_gate, const float* __restrict__ b_gate,
                const _Float16* w_c0, const float* b_c0,
                const float* __restrict__ w_c1, const float* __restrict__ b_c1,
                const float* __restrict__ scale_p,
                float* __restrict__ m_i, float* __restrict__ num,
                float* __restrict__ cnt) {
  __shared__ __align__(32) _Float16 ein_s[16 * 416];   // 13 K-tiles, frag layout
  __shared__ __align__(32) _Float16 mid_s[16 * 800];   // 25 K-tiles, frag layout
  __shared__ __align__(32) _Float16 m_s[16 * 128];     // 4 K-tiles, frag layout
  __shared__ __align__(32) _Float16 coor_s[16 * 512];  // plain row-major (dot only)
  __shared__ float reln_s[16 * 3];
  __shared__ float g_s[16];
  __shared__ int row_s[16], col_s[16];

  const int tid = threadIdx.x, wave = tid >> 5, lane = tid & 31;
  const int e0 = blockIdx.x * 16;

  // ---- Phase A: per-edge metadata + geometry + tail of e_in ----
  if (tid < 16) {
    int e = e0 + tid;
    int rn = eidx[e], cn = eidx[NEDGES + e];
    row_s[tid] = rn; col_s[tid] = cn;
    float dx = posc[rn * 3 + 0] - posc[cn * 3 + 0];
    float dy = posc[rn * 3 + 1] - posc[cn * 3 + 1];
    float dz = posc[rn * 3 + 2] - posc[cn * 3 + 2];
    float d2 = dx * dx + dy * dy + dz * dz;
    float inv = scale_p[0] / (sqrtf(d2) + 1e-8f);
    reln_s[tid * 3 + 0] = dx * inv;
    reln_s[tid * 3 + 1] = dy * inv;
    reln_s[tid * 3 + 2] = dz * inv;
    float tail[8] = {d2, 0, 0, 0, 0, 0, 0, 0};
    store_a8(ein_s, tid, 384, tail);
    float z[8] = {0, 0, 0, 0, 0, 0, 0, 0};
    store_a8(ein_s, tid, 392, z);
    store_a8(ein_s, tid, 400, z);
    store_a8(ein_s, tid, 408, z);
  }
  __syncthreads();
  // gather h[row], h[col], edge_attr in vector groups of 8
  for (int gi = tid; gi < 16 * 48; gi += 128) {        // 384 feats = 48 groups
    int r = gi / 48, f0 = (gi % 48) << 3;
    if (f0 < 256) {
      const float* src = (f0 < 128)
          ? h + (size_t)row_s[r] * 128 + f0
          : h + (size_t)col_s[r] * 128 + (f0 - 128);
      float v[8];
#pragma unroll
      for (int i = 0; i < 8; ++i) v[i] = src[i];
      store_a8(ein_s, r, f0, v);
    } else {
      v8h ev = *(const v8h*)(eattr + (size_t)(e0 + r) * 128 + (f0 - 256));
      *(v8h*)(ein_s + aidx(r, f0)) = ev;
    }
  }
  __syncthreads();

  // ---- Phase B: edge0 (K=13 tiles, N=50 tiles) + SiLU ----
  for (int nt = wave; nt < 50; nt += 4) {
    v8f acc = mma_tile(ein_s, w_e0, 50, b_e0, 770, 13, nt, lane);
    int col = nt * 16 + (lane & 15);
#pragma unroll
    for (int r = 0; r < 8; ++r) {
      float s = (col < 770) ? silu_f(acc[r]) : 0.0f;   // keep edge1 K-pad zero
      mid_s[aidx(frag_m(r, lane), col)] = (_Float16)s;
    }
  }
  __syncthreads();

  // ---- Phase C: edge1 (K=25 tiles, N=8 tiles) + SiLU ----
  for (int nt = wave; nt < 8; nt += 4) {
    v8f acc = mma_tile(mid_s, w_e1, 8, b_e1, 128, 25, nt, lane);
    int col = nt * 16 + (lane & 15);
#pragma unroll
    for (int r = 0; r < 8; ++r)
      m_s[aidx(frag_m(r, lane), col)] = (_Float16)silu_f(acc[r]);
  }
  __syncthreads();

  // soft-edge gate: g = sigmoid(m . w_gate + b)
  if (tid < 16) {
    float a = b_gate[0];
    for (int q = 0; q < 16; ++q) {
      v8h mv = *(const v8h*)(m_s + aidx(tid, q << 3));
#pragma unroll
      for (int i = 0; i < 8; ++i) a += (float)mv[i] * w_gate[(q << 3) + i];
    }
    g_s[tid] = sigmoid_f(a);
  }
  __syncthreads();
  // m *= g ; scatter-add into m_i[row]
  for (int gi = tid; gi < 16 * 16; gi += 128) {
    int r = gi >> 4, f0 = (gi & 15) << 3;
    v8h mv = *(v8h*)(m_s + aidx(r, f0));
    float gv = g_s[r];
    float* dst = m_i + (size_t)row_s[r] * 128 + f0;
#pragma unroll
    for (int i = 0; i < 8; ++i) {
      float x = (float)mv[i] * gv;
      mv[i] = (_Float16)x;
      atomicAdd(&dst[i], x);
    }
    *(v8h*)(m_s + aidx(r, f0)) = mv;
  }
  __syncthreads();

  // ---- Phase D: coor0 (K=4 tiles, N=32 tiles) + SiLU -> plain coor_s ----
  for (int nt = wave; nt < 32; nt += 4) {
    v8f acc = mma_tile(m_s, w_c0, 32, b_c0, 512, 4, nt, lane);
    int col = nt * 16 + (lane & 15);
#pragma unroll
    for (int r = 0; r < 8; ++r)
      coor_s[frag_m(r, lane) * 512 + col] = (_Float16)silu_f(acc[r]);
  }
  __syncthreads();

  // ---- Phase E: coor1 dot + equivariant scatter ----
  if (tid < 16) {
    float a = b_c1[0];
    for (int q = 0; q < 64; ++q) {
      v8h cv = *(const v8h*)(coor_s + tid * 512 + (q << 3));
#pragma unroll
      for (int i = 0; i < 8; ++i) a += (float)cv[i] * w_c1[(q << 3) + i];
    }
    int rn = row_s[tid];
    atomicAdd(&num[rn * 3 + 0], a * reln_s[tid * 3 + 0]);
    atomicAdd(&num[rn * 3 + 1], a * reln_s[tid * 3 + 1]);
    atomicAdd(&num[rn * 3 + 2], a * reln_s[tid * 3 + 2]);
    atomicAdd(&cnt[rn], 1.0f);
  }
}

// ---------------------------------------------------------------------------
// Node update: h += node1(silu(node0([h, m_i]))) ; pos += num/max(cnt,1)
// ---------------------------------------------------------------------------
__global__ void __launch_bounds__(128)
node_update_kernel(float* __restrict__ h, const float* __restrict__ m_i,
                   const _Float16* w_n0, const float* b_n0,
                   const _Float16* w_n1, const float* b_n1,
                   const float* __restrict__ num, const float* __restrict__ cnt,
                   float* __restrict__ posc) {
  __shared__ __align__(32) _Float16 in_s[16 * 256];
  __shared__ __align__(32) _Float16 mid_s[16 * 256];
  const int tid = threadIdx.x, wave = tid >> 5, lane = tid & 31;
  const int n0 = blockIdx.x * 16;
  for (int gi = tid; gi < 16 * 32; gi += 128) {
    int r = gi >> 5, f0 = (gi & 31) << 3;
    const float* src = (f0 < 128) ? h + (size_t)(n0 + r) * 128 + f0
                                  : m_i + (size_t)(n0 + r) * 128 + (f0 - 128);
    float v[8];
#pragma unroll
    for (int i = 0; i < 8; ++i) v[i] = src[i];
    store_a8(in_s, r, f0, v);
  }
  __syncthreads();
  for (int nt = wave; nt < 16; nt += 4) {
    v8f acc = mma_tile(in_s, w_n0, 16, b_n0, 256, 8, nt, lane);
    int col = nt * 16 + (lane & 15);
#pragma unroll
    for (int r = 0; r < 8; ++r)
      mid_s[aidx(frag_m(r, lane), col)] = (_Float16)silu_f(acc[r]);
  }
  __syncthreads();
  for (int nt = wave; nt < 8; nt += 4) {
    v8f acc = mma_tile(mid_s, w_n1, 8, b_n1, 128, 8, nt, lane);
    int col = nt * 16 + (lane & 15);
#pragma unroll
    for (int r = 0; r < 8; ++r)
      h[(size_t)(n0 + frag_m(r, lane)) * 128 + col] += acc[r];  // residual
  }
  if (tid < 16) {
    int n = n0 + tid;
    float c = fmaxf(cnt[n], 1.0f);
    posc[n * 3 + 0] += num[n * 3 + 0] / c;
    posc[n * 3 + 1] += num[n * 3 + 1] / c;
    posc[n * 3 + 2] += num[n * 3 + 2] / c;
  }
}

// ---------------------------------------------------------------------------
// Head: [h[row]*h[col], eattr] -> relu(head0) -> relu(head1) -> head2
// ---------------------------------------------------------------------------
__global__ void __launch_bounds__(128)
head_kernel(const int* __restrict__ eidx, const float* __restrict__ h,
            const _Float16* __restrict__ eattr,
            const _Float16* w_h0, const float* b_h0,
            const _Float16* w_h1, const float* b_h1,
            const float* __restrict__ w_h2, const float* __restrict__ b_h2,
            float* __restrict__ out) {
  __shared__ __align__(32) _Float16 p_s[16 * 256];
  __shared__ __align__(32) _Float16 h1_s[16 * 128];
  __shared__ __align__(32) _Float16 h2_s[16 * 64];   // plain (dot only)
  __shared__ int row_s[16], col_s[16];
  const int tid = threadIdx.x, wave = tid >> 5, lane = tid & 31;
  const int e0 = blockIdx.x * 16;
  if (tid < 16) { row_s[tid] = eidx[e0 + tid]; col_s[tid] = eidx[NEDGES + e0 + tid]; }
  __syncthreads();
  for (int gi = tid; gi < 16 * 32; gi += 128) {
    int r = gi >> 5, f0 = (gi & 31) << 3;
    if (f0 < 128) {
      const float* pa = h + (size_t)row_s[r] * 128 + f0;
      const float* pb = h + (size_t)col_s[r] * 128 + f0;
      float v[8];
#pragma unroll
      for (int i = 0; i < 8; ++i) v[i] = pa[i] * pb[i];
      store_a8(p_s, r, f0, v);
    } else {
      v8h ev = *(const v8h*)(eattr + (size_t)(e0 + r) * 128 + (f0 - 128));
      *(v8h*)(p_s + aidx(r, f0)) = ev;
    }
  }
  __syncthreads();
  for (int nt = wave; nt < 8; nt += 4) {
    v8f acc = mma_tile(p_s, w_h0, 8, b_h0, 128, 8, nt, lane);
    int col = nt * 16 + (lane & 15);
#pragma unroll
    for (int r = 0; r < 8; ++r)
      h1_s[aidx(frag_m(r, lane), col)] = (_Float16)fmaxf(acc[r], 0.0f);
  }
  __syncthreads();
  for (int nt = wave; nt < 4; nt += 4) {
    v8f acc = mma_tile(h1_s, w_h1, 4, b_h1, 64, 4, nt, lane);
    int col = nt * 16 + (lane & 15);
#pragma unroll
    for (int r = 0; r < 8; ++r)
      h2_s[frag_m(r, lane) * 64 + col] = (_Float16)fmaxf(acc[r], 0.0f);
  }
  __syncthreads();
  if (tid < 16) {
    float a = b_h2[0];
    for (int q = 0; q < 8; ++q) {
      v8h cv = *(const v8h*)(h2_s + tid * 64 + (q << 3));
#pragma unroll
      for (int i = 0; i < 8; ++i) a += (float)cv[i] * w_h2[(q << 3) + i];
    }
    out[e0 + tid] = a;
  }
}

// ---------------------------------------------------------------------------
// Host side
// d_in flat layout (jax pytree order: dict keys sorted; each Linear -> {b,w}):
// 0 atom_type 1 pos 2 edge_index 3 edge_type 4 edge_length 5 batch 6 time_step
// 7 bond_emb 8/9 edge_mlp0.b/w 10/11 edge_mlp1.b/w 12/13 embed.b/w
// 14/15 head0.b/w 16/17 head1.b/w 18/19 head2.b/w
// layers i=0..2 at 20+15*i: coor0.b/w coor1.b/w edge0.b/w edge1.b/w gate.b/w
//                            node0.b/w node1.b/w scale
// 65/66 temb0.b/w 67/68 temb1.b/w 69/70 temb_proj.b/w
// ---------------------------------------------------------------------------
extern "C" void kernel_launch(void* const* d_in, const int* in_sizes, int n_in,
                              void* d_out, int out_size, void* d_ws, size_t ws_size,
                              hipStream_t stream) {
  (void)in_sizes; (void)n_in; (void)out_size; (void)ws_size;
  const float* atom_type = (const float*)d_in[0];
  const float* pos       = (const float*)d_in[1];
  const int*   eidx      = (const int*)d_in[2];
  const int*   etype     = (const int*)d_in[3];
  const float* elen      = (const float*)d_in[4];
  const int*   batch     = (const int*)d_in[5];
  const int*   tstep     = (const int*)d_in[6];
  const float* bond_emb  = (const float*)d_in[7];
  float* out = (float*)d_out;

  const int LBASE = 20, LSTR = 15;
  auto Lp = [&](int l, int k) -> const float* { return (const float*)d_in[LBASE + LSTR * l + k]; };

  char* ws = (char*)d_ws;
  size_t off = 0;
  auto alloc = [&](size_t bytes) -> void* {
    void* p = ws + off;
    off = (off + bytes + 255) & ~(size_t)255;
    return p;
  };
  auto fragsz = [](int KT, int NT) -> size_t { return (size_t)KT * NT * 512 * sizeof(_Float16); };

  // f16 fragment-layout weights
  _Float16* wt0  = (_Float16*)alloc(fragsz(4, 32));
  _Float16* wt1  = (_Float16*)alloc(fragsz(16, 32));
  _Float16* wtp  = (_Float16*)alloc(fragsz(16, 8));
  _Float16* wem1 = (_Float16*)alloc(fragsz(4, 8));
  _Float16* wemb = (_Float16*)alloc(fragsz(5, 8));
  _Float16* wh0  = (_Float16*)alloc(fragsz(8, 8));
  _Float16* wh1  = (_Float16*)alloc(fragsz(4, 4));
  _Float16 *we0[3], *we1[3], *wc0[3], *wn0[3], *wn1[3];
  for (int l = 0; l < 3; ++l) {
    we0[l] = (_Float16*)alloc(fragsz(13, 50));
    we1[l] = (_Float16*)alloc(fragsz(25, 8));
    wc0[l] = (_Float16*)alloc(fragsz(4, 32));
    wn0[l] = (_Float16*)alloc(fragsz(8, 16));
    wn1[l] = (_Float16*)alloc(fragsz(8, 8));
  }
  // f32 activation / aggregation buffers
  float* temb  = (float*)alloc(sizeof(float) * NGRAPH * 128);
  float* h     = (float*)alloc(sizeof(float) * (size_t)NNODES * 128);
  float* m_i   = (float*)alloc(sizeof(float) * (size_t)NNODES * 128);
  float* num   = (float*)alloc(sizeof(float) * NNODES * 3);
  float* cnt   = (float*)alloc(sizeof(float) * NNODES);
  float* posc  = (float*)alloc(sizeof(float) * NNODES * 3);
  _Float16* eattr = (_Float16*)alloc(sizeof(_Float16) * (size_t)NEDGES * 128);

  auto cvt = [&](int widx, _Float16* dst, int K, int Nn, int KT, int NT) {
    int tot = KT * NT * 512;
    convert_frag_kernel<<<(tot + 255) / 256, 256, 0, stream>>>(
        (const float*)d_in[widx], dst, K, Nn, KT, NT);
  };
  cvt(66, wt0, 128, 512, 4, 32);
  cvt(68, wt1, 512, 512, 16, 32);
  cvt(70, wtp, 512, 128, 16, 8);
  cvt(11, wem1, 128, 128, 4, 8);
  cvt(13, wemb, NUM_ATOM + HID, 128, 5, 8);
  cvt(15, wh0, 256, 128, 8, 8);
  cvt(17, wh1, 128, 64, 4, 4);
  for (int l = 0; l < 3; ++l) {
    cvt(LBASE + LSTR * l + 5,  we0[l], 385, 770, 13, 50);
    cvt(LBASE + LSTR * l + 7,  we1[l], 770, 128, 25, 8);
    cvt(LBASE + LSTR * l + 1,  wc0[l], 128, 512, 4, 32);
    cvt(LBASE + LSTR * l + 11, wn0[l], 256, 256, 8, 16);
    cvt(LBASE + LSTR * l + 13, wn1[l], 256, 128, 8, 8);
  }

  temb_kernel<<<NGRAPH / 16, 128, 0, stream>>>(
      tstep, wt0, (const float*)d_in[65], wt1, (const float*)d_in[67],
      wtp, (const float*)d_in[69], temb);

  edge_attr_kernel<<<NEDGES / 16, 128, 0, stream>>>(
      elen, etype, eidx, batch,
      (const float*)d_in[9], (const float*)d_in[8],
      wem1, (const float*)d_in[10], bond_emb, temb, eattr);

  embed_kernel<<<NNODES / 16, 128, 0, stream>>>(
      atom_type, batch, temb, wemb, (const float*)d_in[12], h);
  copy_pos_kernel<<<(NNODES * 3 + 255) / 256, 256, 0, stream>>>(pos, posc, NNODES * 3);

  for (int l = 0; l < 3; ++l) {
    zero_f32_kernel<<<(NNODES * 128 + 255) / 256, 256, 0, stream>>>(m_i, NNODES * 128);
    zero_f32_kernel<<<(NNODES * 3 + 255) / 256, 256, 0, stream>>>(num, NNODES * 3);
    zero_f32_kernel<<<(NNODES + 255) / 256, 256, 0, stream>>>(cnt, NNODES);

    edge_msg_kernel<<<NEDGES / 16, 128, 0, stream>>>(
        eidx, posc, h, eattr,
        we0[l], Lp(l, 4), we1[l], Lp(l, 6),
        Lp(l, 9), Lp(l, 8),
        wc0[l], Lp(l, 0), Lp(l, 3), Lp(l, 2),
        Lp(l, 14),
        m_i, num, cnt);

    node_update_kernel<<<NNODES / 16, 128, 0, stream>>>(
        h, m_i, wn0[l], Lp(l, 10), wn1[l], Lp(l, 12), num, cnt, posc);
  }

  head_kernel<<<NEDGES / 16, 128, 0, stream>>>(
      eidx, h, eattr, wh0, (const float*)d_in[14], wh1, (const float*)d_in[16],
      (const float*)d_in[19], (const float*)d_in[18], out);
  aux_out_kernel<<<(NEDGES + 255) / 256, 256, 0, stream>>>(eidx, etype, elen, out);
}
